// MLPBlock_5282809774796
// MI455X (gfx1250) — compile-verified, matrix operand order checked by reading, not run on hardware
//
#include <hip/hip_runtime.h>

// ---------------------------------------------------------------------------
// MoE MLP block (T=2048, H=2048, I=2048, E=16, K=4) for gfx1250 / MI455X.
// Routed (top-4 only), bf16 WMMA with fp32 accumulate, async-LDS staging.
// ---------------------------------------------------------------------------

typedef unsigned short ushort_t;
typedef __attribute__((ext_vector_type(16))) __bf16 v16bf;
typedef __attribute__((ext_vector_type(8)))  float  v8f;
typedef __attribute__((ext_vector_type(4)))  int    v4i;

#define T_TOK 2048
#define H_DIM 2048
#define I_DIM 2048
#define E_NUM 16
#define K_TOP 4
#define ALPHA_C 1.702f
#define LIMIT_C 7.0f

#define TM 128          // tokens per block tile
#define KSTEP 32        // bf16 WMMA K
#define LDS_STRIDE 40   // 32 + 8 pad ushorts -> 80B rows (16B aligned, bank-skewed)

struct U32x8 { uint4 lo, hi; };

__device__ __forceinline__ ushort_t f2bf(float f) {
  unsigned u = __builtin_bit_cast(unsigned, f);
  u += 0x7FFFu + ((u >> 16) & 1u);       // round-to-nearest-even
  return (ushort_t)(u >> 16);
}

// pack two fp32 -> 2x bf16 in one dword: 2x v_add (round bias) + 1x v_perm_b32
__device__ __forceinline__ unsigned pack2bf(float a, float b) {
  unsigned ua = __builtin_bit_cast(unsigned, a) + 0x8000u;   // round-half-up
  unsigned ub = __builtin_bit_cast(unsigned, b) + 0x8000u;
#if __has_builtin(__builtin_amdgcn_perm)
  // result = { ub[31:16], ua[31:16] }  (bytes: 7,6 from s0=ub; 3,2 from s1=ua)
  return __builtin_amdgcn_perm(ub, ua, 0x07060302u);
#else
  return (ua >> 16) | (ub & 0xFFFF0000u);
#endif
}

__device__ __forceinline__ void store4bf(ushort_t* p, float4 v) {
  uint2 u;
  u.x = pack2bf(v.x, v.y);
  u.y = pack2bf(v.z, v.w);
  *(uint2*)p = u;                        // single ds_store_b64
}

// async global->LDS b128 (ASYNCcnt path); falls back to a plain copy.
// Probe-learned signature: (v4i AS1* gsrc, v4i AS3* ldst, imm offset, imm cpol)
__device__ __forceinline__ void async_copy_b128(const ushort_t* g, ushort_t* l) {
#if __has_builtin(__builtin_amdgcn_global_load_async_to_lds_b128)
  __builtin_amdgcn_global_load_async_to_lds_b128(
      (__attribute__((address_space(1))) v4i*)g,
      (__attribute__((address_space(3))) v4i*)l, 0, 0);
#else
  *(uint4*)l = *(const uint4*)g;
#endif
}

__device__ __forceinline__ void wait_async0() {
#if __has_builtin(__builtin_amdgcn_s_wait_asynccnt)
  __builtin_amdgcn_s_wait_asynccnt(0);
#else
  asm volatile("s_wait_asynccnt 0x0" ::: "memory");
#endif
}

// Per-ISA 16-bit operand layout: lane half h holds K = h*8..h*8+7 (elems 0-7)
// and K = 16+h*8..16+h*8+7 (elems 8-15). Two b128 LDS reads per fragment.
__device__ __forceinline__ v16bf frag_from_lds(const ushort_t* rowbase, int halfsel) {
  U32x8 t;
  t.lo = *(const uint4*)(rowbase + halfsel * 8);
  t.hi = *(const uint4*)(rowbase + 16 + halfsel * 8);
  return __builtin_bit_cast(v16bf, t);
}

#define WMMA_BF16(A, B, C) \
  __builtin_amdgcn_wmma_f32_16x16x32_bf16(false, (A), false, (B), (short)0, (C), false, false)

// ---------------------------------------------------------------------------
__global__ __launch_bounds__(256) void zero_kernel(float* __restrict__ out,
                                                   int* __restrict__ counts) {
  int i = blockIdx.x * 256 + threadIdx.x;              // 4096 blocks -> T*H/4
  ((float4*)out)[i] = make_float4(0.f, 0.f, 0.f, 0.f);
  if (blockIdx.x == 0 && threadIdx.x < E_NUM) counts[threadIdx.x] = 0;
}

// ---------------------------------------------------------------------------
__global__ __launch_bounds__(128) void router_kernel(
    const float* __restrict__ x, const float* __restrict__ rw,
    const float* __restrict__ rb, float* __restrict__ combine,
    int* __restrict__ counts, int* __restrict__ lists) {
  __shared__ float xr[H_DIM];
  __shared__ float part[128];
  __shared__ float logit[E_NUM];

  const int t = blockIdx.x;
  const int tid = threadIdx.x;

  for (int i = tid; i < H_DIM / 4; i += 128)
    ((float4*)xr)[i] = ((const float4*)(x + (size_t)t * H_DIM))[i];
  __syncthreads();

  const int e = tid & 15, seg = tid >> 4;              // 8 lanes per expert
  const float* wrow = rw + (size_t)e * H_DIM + seg * 256;
  const float* xrow = xr + seg * 256;
  float s = 0.f;
  for (int i = 0; i < 256; i += 4) {
    float4 w4 = *(const float4*)(wrow + i);
    s += w4.x * xrow[i] + w4.y * xrow[i + 1] + w4.z * xrow[i + 2] + w4.w * xrow[i + 3];
  }
  part[seg * 16 + e] = s;
  __syncthreads();

  if (tid < E_NUM) {
    float l = rb[tid];
    #pragma unroll
    for (int sg = 0; sg < 8; ++sg) l += part[sg * 16 + tid];
    logit[tid] = l;
  }
  __syncthreads();

  if (tid == 0) {
    float v[E_NUM];
    #pragma unroll
    for (int i = 0; i < E_NUM; ++i) v[i] = logit[i];
    int idx[K_TOP]; float val[K_TOP];
    #pragma unroll
    for (int k = 0; k < K_TOP; ++k) {
      int bi = 0; float bv = -3.4e38f;
      #pragma unroll
      for (int i = 0; i < E_NUM; ++i)
        if (v[i] > bv) { bv = v[i]; bi = i; }
      idx[k] = bi; val[k] = bv; v[bi] = -3.4e38f;
    }
    float wv[K_TOP]; float wsum = 0.f;
    #pragma unroll
    for (int k = 0; k < K_TOP; ++k) { wv[k] = __expf(val[k] - val[0]); wsum += wv[k]; }
    float inv = 1.f / wsum;
    float cmb[E_NUM];
    #pragma unroll
    for (int i = 0; i < E_NUM; ++i) cmb[i] = 0.f;
    #pragma unroll
    for (int k = 0; k < K_TOP; ++k) cmb[idx[k]] = wv[k] * inv;
    #pragma unroll
    for (int i = 0; i < E_NUM; ++i) combine[(size_t)t * E_NUM + i] = cmb[i];
    #pragma unroll
    for (int k = 0; k < K_TOP; ++k) {
      int p = atomicAdd(&counts[idx[k]], 1);
      lists[idx[k] * T_TOK + p] = t;
    }
  }
}

// ---------------------------------------------------------------------------
__global__ void scan_kernel(const int* __restrict__ counts, int* __restrict__ offsets) {
  if (threadIdx.x == 0) {
    int a = 0;
    for (int e2 = 0; e2 < E_NUM; ++e2) { offsets[e2] = a; a += counts[e2]; }
    offsets[E_NUM] = a;
  }
}

// ---------------------------------------------------------------------------
// GEMM1: h = gather(x) @ w13[e]^T (+b13), fused clamp/act -> compact bf16 act.
// Block tile: 128 tokens x (64 gate + 64 up). 8 waves; each wave owns a
// 32-row x (32 gate + 32 up matched-j) tile -> activation is register-local.
__global__ __launch_bounds__(256) void gemm1_kernel(
    const float* __restrict__ x, const float* __restrict__ w13,
    const float* __restrict__ b13, const int* __restrict__ counts,
    const int* __restrict__ offsets, const int* __restrict__ lists,
    ushort_t* __restrict__ act) {
  __shared__ alignas(16) ushort_t lds_a[TM * LDS_STRIDE];     // 128 x 32 bf16
  __shared__ alignas(16) ushort_t lds_b[128 * LDS_STRIDE];    // 64 gate + 64 up
  __shared__ int tok[TM];

  const int e = blockIdx.y;
  const int n0 = blockIdx.x * 64;          // gate cols [n0,n0+64), up cols [I+n0,...)
  const int tid = threadIdx.x;
  const int lane = tid & 31;
  const int wave = tid >> 5;
  const int wr = wave & 3;                 // 4 row subtiles (x32)
  const int wc = wave >> 2;                // 2 col subtiles (x32 gate + x32 up)
  const int halfsel = lane >> 4;
  const int l15 = lane & 15;
  const int cnt = counts[e];
  const int off = offsets[e];

  // k-invariant staging geometry (row/chunk per rep)
  int srow[4], schunk[4];
  #pragma unroll
  for (int rep = 0; rep < 4; ++rep) {
    int idx = rep * 256 + tid;
    srow[rep] = idx >> 3; schunk[rep] = idx & 7;
  }

  for (int m0 = 0; m0 < cnt; m0 += TM) {
    __syncthreads();
    if (tid < TM) {
      int m = m0 + tid;
      tok[tid] = lists[e * T_TOK + (m < cnt ? m : cnt - 1)];  // branchless clamp
    }
    __syncthreads();

    v8f ag[2][2] = {}, au[2][2] = {};                          // [mi][ni]
    float4 pa[4], pb[4];

    // hoist k-invariant row pointers, then prefetch k = 0
    const float* arow[4]; const float* brow[4];
    #pragma unroll
    for (int rep = 0; rep < 4; ++rep) {
      int row = srow[rep], c = schunk[rep];
      arow[rep] = x + (size_t)tok[row] * H_DIM + c * 4;
      int o = (row < 64) ? (n0 + row) : (I_DIM + n0 + row - 64);
      brow[rep] = w13 + ((size_t)e * 2 * I_DIM + o) * H_DIM + c * 4;
      pa[rep] = *(const float4*)(arow[rep]);
      pb[rep] = *(const float4*)(brow[rep]);
    }

    for (int k0 = 0; k0 < H_DIM; k0 += KSTEP) {
      // commit prefetched tiles to LDS (fp32 -> bf16)
      #pragma unroll
      for (int rep = 0; rep < 4; ++rep) {
        store4bf(lds_a + srow[rep] * LDS_STRIDE + schunk[rep] * 4, pa[rep]);
        store4bf(lds_b + srow[rep] * LDS_STRIDE + schunk[rep] * 4, pb[rep]);
      }
      __syncthreads();

      if (k0 + KSTEP < H_DIM) {            // prefetch k+1 during compute
        int k1 = k0 + KSTEP;
        #pragma unroll
        for (int rep = 0; rep < 4; ++rep) {
          pa[rep] = *(const float4*)(arow[rep] + k1);
          pb[rep] = *(const float4*)(brow[rep] + k1);
        }
      }

      v16bf a0 = frag_from_lds(lds_a + (wr * 32 +  0 + l15) * LDS_STRIDE, halfsel);
      v16bf a1 = frag_from_lds(lds_a + (wr * 32 + 16 + l15) * LDS_STRIDE, halfsel);
      v16bf g0 = frag_from_lds(lds_b + (wc * 32 +  0 + l15) * LDS_STRIDE, halfsel);
      v16bf g1 = frag_from_lds(lds_b + (wc * 32 + 16 + l15) * LDS_STRIDE, halfsel);
      v16bf u0 = frag_from_lds(lds_b + (64 + wc * 32 +  0 + l15) * LDS_STRIDE, halfsel);
      v16bf u1 = frag_from_lds(lds_b + (64 + wc * 32 + 16 + l15) * LDS_STRIDE, halfsel);
      ag[0][0] = WMMA_BF16(a0, g0, ag[0][0]);
      ag[0][1] = WMMA_BF16(a0, g1, ag[0][1]);
      ag[1][0] = WMMA_BF16(a1, g0, ag[1][0]);
      ag[1][1] = WMMA_BF16(a1, g1, ag[1][1]);
      au[0][0] = WMMA_BF16(a0, u0, au[0][0]);
      au[0][1] = WMMA_BF16(a0, u1, au[0][1]);
      au[1][0] = WMMA_BF16(a1, u0, au[1][0]);
      au[1][1] = WMMA_BF16(a1, u1, au[1][1]);
      __syncthreads();
    }

    // register-local activation epilogue (gate/up share (lane,r) mapping)
    {
      int mbase = wr * 32 + (halfsel ? 8 : 0);
      int jl = wc * 32 + l15;                    // + ni*16
      float bg[2], bu[2];
      #pragma unroll
      for (int ni = 0; ni < 2; ++ni) {
        bg[ni] = b13[e * (2 * I_DIM) + n0 + jl + ni * 16];
        bu[ni] = b13[e * (2 * I_DIM) + I_DIM + n0 + jl + ni * 16];
      }
      #pragma unroll
      for (int mi = 0; mi < 2; ++mi) {
        #pragma unroll
        for (int r = 0; r < 8; ++r) {
          int m = mbase + mi * 16 + r;
          if (m0 + m < cnt) {
            size_t rowoff = (size_t)(off + m0 + m) * I_DIM + n0 + jl;
            #pragma unroll
            for (int ni = 0; ni < 2; ++ni) {
              float gate = ag[mi][ni][r] + bg[ni];
              float up   = au[mi][ni][r] + bu[ni];
              gate = fminf(gate, LIMIT_C);
              up   = fminf(fmaxf(up, -LIMIT_C), LIMIT_C);
              float sg = 1.f / (1.f + __expf(-ALPHA_C * gate));
              act[rowoff + ni * 16] = f2bf(gate * sg * (up + 1.f));
            }
          }
        }
      }
    }
  }
}

// ---------------------------------------------------------------------------
// GEMM2: y = act @ w2[e]^T; epilogue (acc + b2) * combine -> atomic scatter.
// A operand is bf16 in memory: staged via async global->LDS, double-buffered.
__global__ __launch_bounds__(256) void gemm2_kernel(
    const ushort_t* __restrict__ act, const float* __restrict__ w2,
    const float* __restrict__ b2, const float* __restrict__ combine,
    const int* __restrict__ counts, const int* __restrict__ offsets,
    const int* __restrict__ lists, float* __restrict__ out) {
  __shared__ alignas(16) ushort_t lds_a[2][TM * LDS_STRIDE];  // double buffer
  __shared__ alignas(16) ushort_t lds_b[128 * LDS_STRIDE];
  __shared__ int tok[TM];

  const int e = blockIdx.y;
  const int n0 = blockIdx.x * 128;
  const int tid = threadIdx.x;
  const int lane = tid & 31;
  const int wave = tid >> 5;
  const int wr = wave & 3;                 // 4 row subtiles (x32)
  const int wc = wave >> 2;                // 2 col subtiles (x64)
  const int halfsel = lane >> 4;
  const int l15 = lane & 15;
  const int cnt = counts[e];
  const int off = offsets[e];

  for (int m0 = 0; m0 < cnt; m0 += TM) {
    __syncthreads();
    if (tid < TM) {
      int m = m0 + tid;
      tok[tid] = lists[e * T_TOK + (m < cnt ? m : cnt - 1)];
    }
    __syncthreads();

    v8f acc[2][4] = {};                                        // [mi][ni]
    float4 pb[4];

    // hoist k-invariant async src pointers / LDS offsets; B row pointers
    const ushort_t* asrc[2]; int aoff[2];
    #pragma unroll
    for (int rep = 0; rep < 2; ++rep) {
      int idx = rep * 256 + tid, row = idx >> 2, c = idx & 3;
      int m = m0 + row; if (m >= cnt) m = cnt - 1;
      asrc[rep] = act + (size_t)(off + m) * I_DIM + c * 8;
      aoff[rep] = row * LDS_STRIDE + c * 8;
      async_copy_b128(asrc[rep], lds_a[0] + aoff[rep]);        // k = 0
    }
    const float* wrow[4]; int boff[4];
    #pragma unroll
    for (int rep = 0; rep < 4; ++rep) {
      int idx = rep * 256 + tid, row = idx >> 3, c = idx & 7;
      wrow[rep] = w2 + ((size_t)e * H_DIM + n0 + row) * I_DIM + c * 4;
      boff[rep] = row * LDS_STRIDE + c * 4;
      pb[rep] = *(const float4*)(wrow[rep]);                   // k = 0
    }

    for (int s = 0; s < I_DIM / KSTEP; ++s) {
      const int k0 = s * KSTEP;
      const int cur = s & 1;
      // commit B regs to LDS; A arrives via async engine
      #pragma unroll
      for (int rep = 0; rep < 4; ++rep)
        store4bf(lds_b + boff[rep], pb[rep]);
      wait_async0();                       // A(k0) resident in lds_a[cur]
      __syncthreads();

      if (k0 + KSTEP < I_DIM) {            // stage k+1 during compute
        int k1 = k0 + KSTEP;
        #pragma unroll
        for (int rep = 0; rep < 2; ++rep)
          async_copy_b128(asrc[rep] + k1, lds_a[cur ^ 1] + aoff[rep]);
        #pragma unroll
        for (int rep = 0; rep < 4; ++rep)
          pb[rep] = *(const float4*)(wrow[rep] + k1);
      }

      v16bf a0 = frag_from_lds(lds_a[cur] + (wr * 32 +  0 + l15) * LDS_STRIDE, halfsel);
      v16bf a1 = frag_from_lds(lds_a[cur] + (wr * 32 + 16 + l15) * LDS_STRIDE, halfsel);
      #pragma unroll
      for (int ni = 0; ni < 4; ++ni) {
        v16bf bfr = frag_from_lds(lds_b + (wc * 64 + ni * 16 + l15) * LDS_STRIDE, halfsel);
        acc[0][ni] = WMMA_BF16(a0, bfr, acc[0][ni]);
        acc[1][ni] = WMMA_BF16(a1, bfr, acc[1][ni]);
      }
      __syncthreads();
    }

    // weighted atomic scatter epilogue
    {
      int mbase = wr * 32 + (halfsel ? 8 : 0);
      int nb = wc * 64 + l15;
      float b2v[4];
      #pragma unroll
      for (int ni = 0; ni < 4; ++ni) b2v[ni] = b2[e * H_DIM + n0 + nb + ni * 16];
      #pragma unroll
      for (int mi = 0; mi < 2; ++mi) {
        #pragma unroll
        for (int r = 0; r < 8; ++r) {
          int m = mbase + mi * 16 + r;
          if (m0 + m < cnt) {
            int tk = tok[m];
            float cw = combine[(size_t)tk * E_NUM + e];
            float* orow = out + (size_t)tk * H_DIM + n0 + nb;
            #pragma unroll
            for (int ni = 0; ni < 4; ++ni)
              atomicAdd(orow + ni * 16, (acc[mi][ni][r] + b2v[ni]) * cw);
          }
        }
      }
    }
  }
}

// ---------------------------------------------------------------------------
extern "C" void kernel_launch(void* const* d_in, const int* in_sizes, int n_in,
                              void* d_out, int out_size, void* d_ws, size_t ws_size,
                              hipStream_t stream) {
  (void)in_sizes; (void)n_in; (void)out_size; (void)ws_size;
  const float* x   = (const float*)d_in[0];
  const float* rw  = (const float*)d_in[1];
  const float* rb  = (const float*)d_in[2];
  const float* w13 = (const float*)d_in[3];
  const float* b13 = (const float*)d_in[4];
  const float* w2  = (const float*)d_in[5];
  const float* b2  = (const float*)d_in[6];
  float* out = (float*)d_out;

  char* ws = (char*)d_ws;
  float*    combine = (float*)(ws);              // T*E fp32   = 128 KB
  int*      counts  = (int*)(ws + 131072);       // E ints
  int*      offsets = (int*)(ws + 131200);       // E+1 ints
  int*      lists   = (int*)(ws + 131584);       // E*T ints   = 128 KB
  ushort_t* act     = (ushort_t*)(ws + 263168);  // (T*K)*I bf16 = 33.5 MB

  zero_kernel<<<(T_TOK * H_DIM / 4) / 256, 256, 0, stream>>>(out, counts);
  router_kernel<<<T_TOK, 128, 0, stream>>>(x, rw, rb, combine, counts, lists);
  scan_kernel<<<1, 32, 0, stream>>>(counts, offsets);
  gemm1_kernel<<<dim3(I_DIM / 64, E_NUM), 256, 0, stream>>>(x, w13, b13, counts, offsets, lists, act);
  gemm2_kernel<<<dim3(H_DIM / 128, E_NUM), 256, 0, stream>>>(act, w2, b2, combine, counts, offsets, lists, out);
}
